// LongConv_9534827397190
// MI455X (gfx1250) — matrix operand order repaired and checked by reading, not compiled, not run on hardware
//
#include <hip/hip_runtime.h>
#include <hip/hip_bf16.h>
#include <math.h>

// ---------------------------------------------------------------------------
// LongConv (Hyena-style) for MI455X / gfx1250, wave32, WMMA bf16 + TDM.
//   y = causal_conv(u, k) + D*u ; y = gelu(y) ; z = W@y + b ; out = GLU(z)
// Conv = implicit GEMM vs. the Toeplitz matrix of k (v_wmma_f32_16x16x32_bf16,
// f32 accumulate), branch-free reversed-kernel operand build.
// Channel GEMM + GLU fused; y tiles staged into LDS by the Tensor Data Mover
// (tensor_load_to_lds, double-buffered, s_wait_tensorcnt).
// ---------------------------------------------------------------------------

typedef __bf16 v16bf  __attribute__((ext_vector_type(16)));
typedef __bf16 bf16x8 __attribute__((ext_vector_type(8)));
typedef __bf16 bf2    __attribute__((ext_vector_type(2)));
typedef float  v8f    __attribute__((ext_vector_type(8)));
typedef unsigned int u32x4 __attribute__((ext_vector_type(4)));
typedef int i32x8 __attribute__((ext_vector_type(8)));
typedef int i32x4 __attribute__((ext_vector_type(4)));

#define B_   8
#define H_   768
#define L_   4096
#define NJ   (L_ / 16)         // 256 column blocks of 16
#define KREV (L_ + 64)         // reversed kernel + zero tail pad
#define NK   (H_ / 32)         // 24 K-steps in the channel GEMM

// ------------------------- kernel 0: W f32 -> bf16 -------------------------
__global__ void wconv_kernel(const float* __restrict__ W,
                             __bf16* __restrict__ Wb, int n) {
    int i = blockIdx.x * blockDim.x + threadIdx.x;
    if (i < n) Wb[i] = (__bf16)W[i];
}

// --------------- kernel 1: Toeplitz-WMMA conv + D skip + GELU --------------
// One workgroup per h; u[:,h,:] + reversed k staged ONCE. 8 waves x 16 passes,
// each pass: wave owns blocks (j0, j0+1), j0 = 16*pass + 2*wave (even), which
// share the same K-range t = 0..j0/2 and the same A tiles.
__global__ __launch_bounds__(256)
void conv_kernel(const float* __restrict__ u, const float* __restrict__ kker,
                 const float* __restrict__ Dv, __bf16* __restrict__ ybf) {
    extern __shared__ char smem[];
    __bf16* u_lds = (__bf16*)smem;                            // [8][4096] 64 KB
    __bf16* krev  = (__bf16*)(smem + B_ * L_ * 2);            // [KREV]
    __bf16* krevo = krev + KREV;                              // [KREV] (shift 1)

    const int h   = blockIdx.x;
    const int tid = threadIdx.x;
    const __bf16 zero = (__bf16)0.0f;

    // Stage u[:,h,:] as bf16 (vectorized: float4 in, bf16x4 out).
    typedef float f4 __attribute__((ext_vector_type(4)));
    typedef __bf16 bf4 __attribute__((ext_vector_type(4)));
    for (int i = tid; i < (B_ * L_) / 4; i += 256) {
        int idx = i * 4;
        int b = idx >> 12, l = idx & (L_ - 1);
        f4 v = *(const f4*)(u + ((size_t)b * H_ + h) * L_ + l);
        bf4 w = { (__bf16)v.x, (__bf16)v.y, (__bf16)v.z, (__bf16)v.w };
        *(bf4*)(u_lds + idx) = w;
    }
    // Stage reversed kernel (+64 zero pad) and its 1-shifted aligned twin.
    for (int x = tid; x < KREV; x += 256) {
        krev[x]  = (x < L_)     ? (__bf16)kker[h * L_ + (L_ - 1 - x)] : zero;
        krevo[x] = (x + 1 < L_) ? (__bf16)kker[h * L_ + (L_ - 2 - x)] : zero;
    }
    __syncthreads();

    const float Dh = Dv[h];
    const int wave = __builtin_amdgcn_readfirstlane(tid >> 5); // wave-uniform
    const int lane = tid & 31;
    const int half = lane >> 4;
    const int p    = lane & 15;       // A row m / C col n
    const int mb   = p & 7;           // batch (A rows 8..15 duplicate)
    const int sbase = (L_ - 1) - p + 16 * half;   // per-lane Toeplitz base

    for (int pass = 0; pass < NJ / 16; ++pass) {
        const int j0   = pass * 16 + 2 * wave;    // even, wave-uniform
        const int l0   = j0 * 16;
        const int tmax = j0 >> 1;                 // scalar loop bound
        v8f acc0 = {}, acc1 = {};
        for (int t = 0; t <= tmax; ++t) {
            // A tile (shared by both output blocks): two aligned b128 loads.
            const __bf16* urow = u_lds + mb * L_ + 32 * t;
            bf16x8 a0 = *(const bf16x8*)(urow + 16 * half);
            bf16x8 a1 = *(const bf16x8*)(urow + 16 * half + 8);
            v16bf av = __builtin_shufflevector(a0, a1,
                0,1,2,3,4,5,6,7,8,9,10,11,12,13,14,15);
            // B tiles: branch-free reversed-kernel reads, parity-aligned.
            const int s0 = sbase - l0 + 32 * t;           // block j0
            const int s1 = s0 - 16;                        // block j0+1
            const unsigned int* ps0 = (const unsigned int*)
                ((s0 & 1) ? (krevo + (s0 - 1)) : (krev + s0));
            const unsigned int* ps1 = (const unsigned int*)
                ((s1 & 1) ? (krevo + (s1 - 1)) : (krev + s1));
            v16bf bv0, bv1;
            #pragma unroll
            for (int d = 0; d < 8; ++d) {
                bf2 q0 = ((const bf2*)ps0)[d];
                bf2 q1 = ((const bf2*)ps1)[d];
                bv0[2*d] = q0[0]; bv0[2*d+1] = q0[1];
                bv1[2*d] = q1[0]; bv1[2*d+1] = q1[1];
            }
            acc0 = __builtin_amdgcn_wmma_f32_16x16x32_bf16(
                false, av, false, bv0, (short)0, acc0, false, false);
            acc1 = __builtin_amdgcn_wmma_f32_16x16x32_bf16(
                false, av, false, bv1, (short)0, acc1, false, false);
        }
        // Epilogue: D rows 0..7 valid -> lanes 0..15 store both blocks.
        if (half == 0) {
            #pragma unroll
            for (int r = 0; r < 8; ++r) {
                int l = l0 + p;
                float x0 = acc0[r] + Dh * (float)u_lds[r * L_ + l];
                float x1 = acc1[r] + Dh * (float)u_lds[r * L_ + l + 16];
                x0 = 0.5f * x0 * (1.0f + erff(x0 * 0.70710678118654752f));
                x1 = 0.5f * x1 * (1.0f + erff(x1 * 0.70710678118654752f));
                __bf16* yr = ybf + ((size_t)r * H_ + h) * L_ + l;
                yr[0]  = (__bf16)x0;
                yr[16] = (__bf16)x1;
            }
        }
    }
}

// ---------------- kernel 2: z = W@gelu(y)+b, GLU, f32 out ------------------
// WG = (b, l-tile); 8 waves cover 8 o-pairs (o, o+H share the y tile -> GLU
// fuses in-register). y tiles (32x16 bf16) are DMA'd into LDS by the Tensor
// Data Mover, double-buffered: tile kk+1 is in flight while kk computes.
__global__ __launch_bounds__(256)
void gemm_glu_kernel(const __bf16* __restrict__ ybf,
                     const __bf16* __restrict__ Wb,
                     const float* __restrict__ bias,
                     float* __restrict__ out) {
    // Only LDS object in this kernel -> static LDS offset 0 (used in the D#).
    __shared__ __bf16 tile[2][32 * 16];            // 2 x 1 KB ping-pong

    const int bid = blockIdx.x;
    const int b   = bid / (NJ * 6);
    const int rem = bid % (NJ * 6);
    const int lt  = rem / 6;
    const int ch  = rem % 6;
    const int tid  = threadIdx.x;
    const int wave = __builtin_amdgcn_readfirstlane(tid >> 5);
    const int lane = tid & 31;
    const int half = lane >> 4, p = lane & 15;

    const int oa0 = (ch * 8 + wave) * 16;     // "a" rows; "g" rows = oa0 + H
    const int l0  = lt * 16;
    const int m   = p;

    // --- TDM descriptor group 1 (constant): 2D bf16 tile 16(x) x 32(y) ---
    i32x8 g1;
    g1[0] = 0x10000;            // workgroup_mask=0, data_size=1 (2 bytes)
    g1[1] = (int)(L_ << 16);    // tensor_dim0 = 4096 (low 16 in [31:16])
    g1[2] = (int)(H_ << 16);    // tensor_dim0 hi=0 | tensor_dim1 = 768
    g1[3] = (int)(16 << 16);    // tensor_dim1 hi=0 | tile_dim0 = 16
    g1[4] = 32;                 // tile_dim1 = 32, tile_dim2 = 0
    g1[5] = L_;                 // tensor_dim0_stride = 4096 elements
    g1[6] = 0;
    g1[7] = 0;
    const i32x4 gz4 = {0, 0, 0, 0};
    const i32x8 gz8 = {0, 0, 0, 0, 0, 0, 0, 0};
    const __bf16* ybase = ybf + (size_t)b * H_ * L_ + l0;

    v8f accA = {}, accG = {};
    const __bf16* wra = Wb + (size_t)(oa0 + m) * H_;
    const __bf16* wrg = Wb + (size_t)(oa0 + H_ + m) * H_;

    // Prologue: DMA tile 0 into buffer 0 (wave 0 only; branch is scalar).
    if (wave == 0) {
        unsigned long long ga = (unsigned long long)(const void*)(ybase);
        u32x4 g0;
        g0[0] = 1u;                                   // count=1, user mode
        g0[1] = 0u;                                   // lds_addr = offset 0
        g0[2] = (unsigned)(ga & 0xffffffffu);
        g0[3] = (unsigned)((ga >> 32) & 0x1ffffffu) | (2u << 30); // type=2
        __builtin_amdgcn_tensor_load_to_lds(g0, g1, gz4, gz4, gz8, 0);
    }

    for (int kk = 0; kk < NK; ++kk) {
        const int k0 = kk * 32;
        if (wave == 0) {
            if (kk + 1 < NK) {   // launch next tile, then wait for current
                unsigned long long ga = (unsigned long long)(const void*)
                    (ybase + (size_t)(k0 + 32) * L_);
                u32x4 g0;
                g0[0] = 1u;
                g0[1] = (unsigned)(((kk + 1) & 1) * (32 * 16 * 2)); // ping-pong
                g0[2] = (unsigned)(ga & 0xffffffffu);
                g0[3] = (unsigned)((ga >> 32) & 0x1ffffffu) | (2u << 30);
                __builtin_amdgcn_tensor_load_to_lds(g0, g1, gz4, gz4, gz8, 0);
                __builtin_amdgcn_s_wait_tensorcnt(1);
            } else {
                __builtin_amdgcn_s_wait_tensorcnt(0);
            }
        }
        __syncthreads();                       // tile kk visible to all waves

        // B operand from row-major LDS tile: k = 8*half+e (+8 for e>=8), n = p.
        const __bf16* lb = &tile[kk & 1][0];
        v16bf bv;
        #pragma unroll
        for (int e = 0; e < 16; ++e)
            bv[e] = lb[(8 * half + e + (e >= 8 ? 8 : 0)) * 16 + p];

        // A operands: two aligned b128 global loads per matrix (+ prefetch).
        bf16x8 aa0 = *(const bf16x8*)(wra + k0 + 8 * half);
        bf16x8 aa1 = *(const bf16x8*)(wra + k0 + 16 + 8 * half);
        bf16x8 ag0 = *(const bf16x8*)(wrg + k0 + 8 * half);
        bf16x8 ag1 = *(const bf16x8*)(wrg + k0 + 16 + 8 * half);
        if (k0 + 32 < H_) {
            __builtin_prefetch(wra + k0 + 32, 0, 1);
            __builtin_prefetch(wrg + k0 + 32, 0, 1);
        }
        v16bf avA = __builtin_shufflevector(aa0, aa1,
            0,1,2,3,4,5,6,7,8,9,10,11,12,13,14,15);
        v16bf avG = __builtin_shufflevector(ag0, ag1,
            0,1,2,3,4,5,6,7,8,9,10,11,12,13,14,15);

        accA = __builtin_amdgcn_wmma_f32_16x16x32_bf16(
            false, avA, false, bv, (short)0, accA, false, false);
        accG = __builtin_amdgcn_wmma_f32_16x16x32_bf16(
            false, avG, false, bv, (short)0, accG, false, false);

        __syncthreads();   // all reads done before the in-flight DMA's buffer
    }                      // (issued next iteration) can be overwritten

    // Epilogue: bias + GLU. D layout: element r -> M = r + 8*half, N = p.
    #pragma unroll
    for (int r = 0; r < 8; ++r) {
        int mo = r + 8 * half;
        float za = accA[r] + bias[oa0 + mo];
        float zg = accG[r] + bias[H_ + oa0 + mo];
        float s  = 1.0f / (1.0f + __expf(-zg));
        out[((size_t)b * H_ + (oa0 + mo)) * L_ + l0 + p] = za * s;
    }
}

// ---------------------------------------------------------------------------
extern "C" void kernel_launch(void* const* d_in, const int* in_sizes, int n_in,
                              void* d_out, int out_size, void* d_ws, size_t ws_size,
                              hipStream_t stream) {
    (void)in_sizes; (void)n_in; (void)out_size; (void)ws_size;
    const float* u    = (const float*)d_in[0];   // (B, H, L)
    const float* k    = (const float*)d_in[1];   // (1, H, L)
    const float* Dv   = (const float*)d_in[2];   // (1, H)
    const float* W    = (const float*)d_in[3];   // (2H, H)
    const float* bias = (const float*)d_in[4];   // (2H,)
    float* out = (float*)d_out;                  // (B, H, L) f32

    // Workspace: [ gelu(y) bf16 : B*H*L ][ W bf16 : 2H*H ]
    __bf16* ybf = (__bf16*)d_ws;
    __bf16* Wb  = (__bf16*)((char*)d_ws + (size_t)B_ * H_ * L_ * 2);

    const int wn = 2 * H_ * H_;
    wconv_kernel<<<(wn + 255) / 256, 256, 0, stream>>>(W, Wb, wn);

    const size_t smem1 = (size_t)B_ * L_ * 2 + (size_t)2 * KREV * 2; // ~80.3 KB
    conv_kernel<<<H_, 256, smem1, stream>>>(u, k, Dv, ybf);

    gemm_glu_kernel<<<B_ * NJ * 6, 256, 0, stream>>>(ybf, Wb, bias, out);
}
